// CausalSelfAttention_90383291777074
// MI455X (gfx1250) — matrix operand order repaired and checked by reading, not compiled, not run on hardware
//
#include <hip/hip_runtime.h>

// ---------------------------------------------------------------------------
// Problem constants (match reference)
// ---------------------------------------------------------------------------
#define Bq   2
#define Tq   2048
#define Cq   1024
#define Hq   16
#define Dq   64
#define Mrows (Bq * Tq)        // 4096
#define N1   (3 * Cq)          // 3072

typedef __bf16 bf16_t;
typedef __attribute__((ext_vector_type(8)))  __bf16 v8bf;
typedef __attribute__((ext_vector_type(16))) __bf16 v16bf;
typedef __attribute__((ext_vector_type(8)))  float  v8f;
typedef __attribute__((ext_vector_type(4)))  unsigned int v4u;
typedef __attribute__((ext_vector_type(8)))  int v8i;
typedef __attribute__((ext_vector_type(4)))  int v4i;

// LDS B-slab: 64 rows, padded row stride 72 bf16 (TDM pad: 4 DW per 32 DW row)
#define BSLAB_LD   72
#define BSLAB_ELEM (64 * BSLAB_LD + 16)

// ---------------------------------------------------------------------------
// Helpers
// ---------------------------------------------------------------------------
__device__ __forceinline__ bf16_t f2bf(float f) {
  unsigned u = __builtin_bit_cast(unsigned, f);
  unsigned r = u + 0x7FFFu + ((u >> 16) & 1u);          // round-to-nearest-even
  unsigned short h = (unsigned short)(r >> 16);
  return __builtin_bit_cast(bf16_t, h);
}

// A-matrix fragment, 16x32 bf16, row-major source (ld in elements).
// ISA 7.12.2: lanes 0-15: row = lane, K = {0..7, 16..23}; lanes 16-31: K = {8..15, 24..31}.
__device__ __forceinline__ v16bf load_a_frag(const bf16_t* __restrict__ base, int ld) {
  int lane = threadIdx.x & 31;
  const bf16_t* p = base + (lane & 15) * ld + ((lane >> 4) << 3);
  v8bf lo = *(const v8bf*)(p);
  v8bf hi = *(const v8bf*)(p + 16);
  v16bf r;
#pragma unroll
  for (int i = 0; i < 8; ++i) { r[i] = lo[i]; r[i + 8] = hi[i]; }
  return r;
}

// B-matrix fragment, 32x16 bf16, from an N-major (transposed, [N][K]) source.
// B pattern matching A: lane l%16 = column N; lanes 0-15 hold K={0..7,16..23}, lanes 16-31 hold K={8..15,24..31}.
__device__ __forceinline__ v16bf load_b_frag(const bf16_t* __restrict__ base, int ld) {
  int lane = threadIdx.x & 31;
  const bf16_t* p = base + (lane & 15) * ld + ((lane >> 4) << 3);
  v8bf lo = *(const v8bf*)(p);
  v8bf hi = *(const v8bf*)(p + 16);
  v16bf r;
#pragma unroll
  for (int i = 0; i < 8; ++i) { r[i] = lo[i]; r[i + 8] = hi[i]; }
  return r;
}

__device__ __forceinline__ v8f wmma_bf16(v16bf a, v16bf b, v8f c) {
  return __builtin_amdgcn_wmma_f32_16x16x32_bf16(false, a, false, b, (short)0, c,
                                                 false, false);
}

// ---------------------------------------------------------------------------
// Tensor Data Mover: stage a 64(N-rows) x 64(K) bf16 tile from an [N][K]
// row-major global matrix (row stride = K elements) into LDS, with D# padding
// of 4 DWORDs every 32 DWORDs -> LDS row stride 72 bf16 (bank-friendly,
// 16B-aligned rows).  D# per CDNA5 ISA (group0 + group1).
// Toolchain uses the 6-arg builtin: (v4u, v8i, v4i, v4i, v8i, i32 cpol).
// ---------------------------------------------------------------------------
__device__ __forceinline__ void tdm_load_b_slab(unsigned lds_off,
                                                const bf16_t* __restrict__ gptr,
                                                int K) {
  unsigned long long ga = (unsigned long long)(size_t)gptr;
  v4u g0;
  g0[0] = 1u;                                     // count=1, user descriptor
  g0[1] = lds_off;                                // lds_addr (bytes)
  g0[2] = (unsigned)(ga & 0xFFFFFFFFu);           // global_addr[31:0]
  g0[3] = (unsigned)(ga >> 32) | 0x80000000u;     // global_addr[56:32] | type=2
  v8i g1;
  // data_size=1 (2B) @[17:16]; pad_enable @[20]; pad_interval=4 (32 DW) @[24:22];
  // pad_amount=3 (4 DW) @[31:25]
  g1[0] = (int)0x07110000u;
  g1[1] = (int)(((unsigned)K & 0xFFFFu) << 16);   // tensor_dim0[15:0] @ bits[63:48]
  g1[2] = (int)((((unsigned)K >> 16) & 0xFFFFu)   // tensor_dim0[31:16]
                | (64u << 16));                   // tensor_dim1[15:0] = 64
  g1[3] = (int)(64u << 16);                       // tensor_dim1[31:16]=0 | tile_dim0=64
  g1[4] = 64;                                     // tile_dim1=64, tile_dim2=0
  g1[5] = K;                                      // tensor_dim0_stride[31:0]
  g1[6] = 0;                                      // stride hi / tensor_dim1_stride lo
  g1[7] = 0;
  v4i gz4 = {0, 0, 0, 0};                         // groups 2/3: tile_dim3/4 = 0 (2D)
  v8i gz8 = {0, 0, 0, 0, 0, 0, 0, 0};             // extra group (clang-23 form): zero
  __builtin_amdgcn_tensor_load_to_lds(g0, g1, gz4, gz4, gz8, 0);
}

// ---------------------------------------------------------------------------
// Prep kernels: f32 -> bf16 conversion / weight transpose
// ---------------------------------------------------------------------------
__global__ void __launch_bounds__(256) csa_cvt_bf16(const float* __restrict__ in,
                                                    bf16_t* __restrict__ out, int n) {
  int i = blockIdx.x * 256 + threadIdx.x;
  if (i < n) out[i] = f2bf(in[i]);
}

// w: [K][N] f32  ->  wt: [N][K] bf16
__global__ void __launch_bounds__(256) csa_transpose_w(const float* __restrict__ w,
                                                       bf16_t* __restrict__ wt,
                                                       int K, int N) {
  int i = blockIdx.x * 256 + threadIdx.x;
  if (i < K * N) {
    int n = i / K, k = i - n * K;
    wt[i] = f2bf(w[k * N + n]);
  }
}

// ---------------------------------------------------------------------------
// Cooperative GEMM mainloop: block tile 128(M) x 64(N), 8 waves (16 rows each).
// B k-slabs (64x64) staged in LDS by wave 0 via TDM, double-buffered.
// ---------------------------------------------------------------------------
__device__ __forceinline__ void gemm_mainloop(const bf16_t* __restrict__ arow,
                                              const bf16_t* __restrict__ bbase,
                                              bf16_t* __restrict__ bsl0,
                                              bf16_t* __restrict__ bsl1,
                                              int K, v8f (&acc)[4]) {
  const int wave = threadIdx.x >> 5;
  const unsigned lds0 = (unsigned)(size_t)(void*)bsl0;
  const unsigned lds1 = (unsigned)(size_t)(void*)bsl1;

  if (wave == 0) tdm_load_b_slab(lds0, bbase, K);

  for (int k0 = 0; k0 < K; k0 += 64) {
    const bf16_t* lb = (k0 & 64) ? bsl1 : bsl0;
    if (wave == 0) {
      if (k0 + 64 < K) {
        tdm_load_b_slab((k0 & 64) ? lds0 : lds1, bbase + (k0 + 64), K);
        __builtin_amdgcn_s_wait_tensorcnt(1);     // current slab landed
      } else {
        __builtin_amdgcn_s_wait_tensorcnt(0);
      }
    }
    __syncthreads();                              // slab visible to all waves
#pragma unroll
    for (int c = 0; c < 2; ++c) {
      v16bf a = load_a_frag(arow + k0 + c * 32, K);
#pragma unroll
      for (int ns = 0; ns < 4; ++ns) {
        v16bf b = load_b_frag(lb + (ns * 16) * BSLAB_LD + c * 32, BSLAB_LD);
        acc[ns] = wmma_bf16(a, b, acc[ns]);
      }
    }
    __syncthreads();                              // all reads done before reuse
  }
}

// ---------------------------------------------------------------------------
// GEMM1: qkv = x @ W_attn + b_attn    (4096 x 3072 x 1024)
// Scatters q,k -> [B,H,T,D] bf16, v -> [B,H,D,T] bf16.
// ---------------------------------------------------------------------------
__global__ void __launch_bounds__(256) csa_gemm_qkv(const bf16_t* __restrict__ xb,
                                                    const bf16_t* __restrict__ wat,
                                                    const float* __restrict__ bias,
                                                    bf16_t* __restrict__ qws,
                                                    bf16_t* __restrict__ kws,
                                                    bf16_t* __restrict__ vtws) {
  __shared__ __align__(16) bf16_t bsl[2][BSLAB_ELEM];
  const int wave = threadIdx.x >> 5, lane = threadIdx.x & 31;
  const int nblk = N1 / 64;                       // 48
  const int bm = blockIdx.x / nblk, bn = blockIdx.x - bm * nblk;
  const int m0 = bm * 128 + wave * 16, n0 = bn * 64;

  v8f z = {0.f, 0.f, 0.f, 0.f, 0.f, 0.f, 0.f, 0.f};
  v8f acc[4] = {z, z, z, z};

  gemm_mainloop(xb + (size_t)m0 * Cq, wat + (size_t)n0 * Cq,
                &bsl[0][0], &bsl[1][0], Cq, acc);

  const int hi8 = (lane >> 4) << 3;
  const int col = lane & 15;
#pragma unroll
  for (int ns = 0; ns < 4; ++ns) {
    const int n = n0 + ns * 16 + col;
    const float bv = bias[n];
#pragma unroll
    for (int i = 0; i < 8; ++i) {
      const int m = m0 + i + hi8;
      const bf16_t o = f2bf(acc[ns][i] + bv);
      const int bb = m >> 11, t = m & (Tq - 1);
      if (n < Cq) {
        const int h = n >> 6, d = n & 63;
        qws[(((size_t)(bb * Hq + h) * Tq) + t) * Dq + d] = o;
      } else if (n < 2 * Cq) {
        const int nn = n - Cq, h = nn >> 6, d = nn & 63;
        kws[(((size_t)(bb * Hq + h) * Tq) + t) * Dq + d] = o;
      } else {
        const int nn = n - 2 * Cq, h = nn >> 6, d = nn & 63;
        vtws[(((size_t)(bb * Hq + h) * Dq) + d) * Tq + t] = o;
      }
    }
  }
}

// ---------------------------------------------------------------------------
// Flash attention: one wave per 16-query tile, key blocks of 64, online softmax.
// q,k: [B,H,T,D] bf16; vt: [B,H,D,T] bf16; y out: [B*T, C] bf16.
// ---------------------------------------------------------------------------
__global__ void __launch_bounds__(256) csa_attention(const bf16_t* __restrict__ qws,
                                                     const bf16_t* __restrict__ kws,
                                                     const bf16_t* __restrict__ vtws,
                                                     bf16_t* __restrict__ yws) {
  __shared__ bf16_t lds_p[8][16 * 72];            // 16 rows x 64 cols, stride 72
  const int wave = threadIdx.x >> 5, lane = threadIdx.x & 31;
  const int tile = blockIdx.x * 8 + wave;         // 4096 tiles
  const int bh = tile >> 7;                       // /(T/16)
  const int tq0 = (tile & 127) << 4;
  const int b = bh >> 4, h = bh & 15;

  const bf16_t* qb = qws + (size_t)bh * Tq * Dq;
  const bf16_t* kb = kws + (size_t)bh * Tq * Dq;
  const bf16_t* vb = vtws + (size_t)bh * Dq * Tq;

  const v16bf qa0 = load_a_frag(qb + (size_t)tq0 * Dq + 0, Dq);
  const v16bf qa1 = load_a_frag(qb + (size_t)tq0 * Dq + 32, Dq);

  v8f z = {0.f, 0.f, 0.f, 0.f, 0.f, 0.f, 0.f, 0.f};
  v8f yacc[4] = {z, z, z, z};
  float mrow[8], lrow[8];
#pragma unroll
  for (int i = 0; i < 8; ++i) { mrow[i] = -__builtin_inff(); lrow[i] = 0.f; }

  const float qk_scale = 0.125f * 1.44269504088896340736f;  // 1/sqrt(64) * log2(e)
  const int hi8 = (lane >> 4) << 3;
  const int col = lane & 15;
  bf16_t* pl = &lds_p[wave][0];
  const int kend = tq0 + 16;

  for (int kb0 = 0; kb0 < kend; kb0 += 64) {
    // ---- S = Q K^T (scaled, base-2 domain) ----
    v8f sacc[4] = {z, z, z, z};
#pragma unroll
    for (int ns = 0; ns < 4; ++ns) {
      v16bf kf0 = load_b_frag(kb + (size_t)(kb0 + ns * 16) * Dq + 0, Dq);
      v16bf kf1 = load_b_frag(kb + (size_t)(kb0 + ns * 16) * Dq + 32, Dq);
      sacc[ns] = wmma_bf16(qa0, kf0, sacc[ns]);
      sacc[ns] = wmma_bf16(qa1, kf1, sacc[ns]);
    }
    // ---- scale + causal mask ----
#pragma unroll
    for (int ns = 0; ns < 4; ++ns) {
      const int tk = kb0 + ns * 16 + col;
#pragma unroll
      for (int i = 0; i < 8; ++i) {
        const int tq = tq0 + i + hi8;
        const float s = sacc[ns][i] * qk_scale;
        sacc[ns][i] = (tk <= tq) ? s : -__builtin_inff();
      }
    }
    // ---- online softmax (rows live across 16-lane C-layout groups) ----
#pragma unroll
    for (int i = 0; i < 8; ++i) {
      float bm = fmaxf(fmaxf(sacc[0][i], sacc[1][i]), fmaxf(sacc[2][i], sacc[3][i]));
#pragma unroll
      for (int off = 1; off < 16; off <<= 1) bm = fmaxf(bm, __shfl_xor(bm, off, 32));
      const float mnew = fmaxf(mrow[i], bm);
      const float corr = __builtin_amdgcn_exp2f(mrow[i] - mnew);
      float psum = 0.f;
#pragma unroll
      for (int ns = 0; ns < 4; ++ns) {
        const float p = __builtin_amdgcn_exp2f(sacc[ns][i] - mnew);
        pl[(i + hi8) * 72 + ns * 16 + col] = f2bf(p);
        psum += p;
      }
#pragma unroll
      for (int off = 1; off < 16; off <<= 1) psum += __shfl_xor(psum, off, 32);
      lrow[i] = lrow[i] * corr + psum;
      mrow[i] = mnew;
#pragma unroll
      for (int ds = 0; ds < 4; ++ds) yacc[ds][i] *= corr;
    }
    // ---- Y += P V  (P re-fetched from LDS in A-fragment layout) ----
#pragma unroll
    for (int c = 0; c < 2; ++c) {
      v16bf pa = load_a_frag(pl + c * 32, 72);
#pragma unroll
      for (int ds = 0; ds < 4; ++ds) {
        v16bf vf = load_b_frag(vb + (size_t)(ds * 16) * Tq + kb0 + c * 32, Tq);
        yacc[ds] = wmma_bf16(pa, vf, yacc[ds]);
      }
    }
  }

  // ---- normalize and write y as [B*T, C] bf16 ----
#pragma unroll
  for (int i = 0; i < 8; ++i) {
    const float rinv = __builtin_amdgcn_rcpf(lrow[i]);
    const int row = b * Tq + tq0 + i + hi8;
#pragma unroll
    for (int ds = 0; ds < 4; ++ds) {
      yws[(size_t)row * Cq + h * 64 + ds * 16 + col] = f2bf(yacc[ds][i] * rinv);
    }
  }
}

// ---------------------------------------------------------------------------
// GEMM2: out = y @ W_proj + b_proj   (4096 x 1024 x 1024), f32 output
// ---------------------------------------------------------------------------
__global__ void __launch_bounds__(256) csa_gemm_proj(const bf16_t* __restrict__ yb,
                                                     const bf16_t* __restrict__ wpt,
                                                     const float* __restrict__ bias,
                                                     float* __restrict__ out) {
  __shared__ __align__(16) bf16_t bsl[2][BSLAB_ELEM];
  const int wave = threadIdx.x >> 5, lane = threadIdx.x & 31;
  const int nblk = Cq / 64;                       // 16
  const int bm = blockIdx.x / nblk, bn = blockIdx.x - bm * nblk;
  const int m0 = bm * 128 + wave * 16, n0 = bn * 64;

  v8f z = {0.f, 0.f, 0.f, 0.f, 0.f, 0.f, 0.f, 0.f};
  v8f acc[4] = {z, z, z, z};

  gemm_mainloop(yb + (size_t)m0 * Cq, wpt + (size_t)n0 * Cq,
                &bsl[0][0], &bsl[1][0], Cq, acc);

  const int hi8 = (lane >> 4) << 3;
  const int col = lane & 15;
#pragma unroll
  for (int ns = 0; ns < 4; ++ns) {
    const int n = n0 + ns * 16 + col;
    const float bv = bias[n];
#pragma unroll
    for (int i = 0; i < 8; ++i) {
      const int m = m0 + i + hi8;
      out[(size_t)m * Cq + n] = acc[ns][i] + bv;
    }
  }
}

// ---------------------------------------------------------------------------
// Host launcher
// ---------------------------------------------------------------------------
extern "C" void kernel_launch(void* const* d_in, const int* in_sizes, int n_in,
                              void* d_out, int out_size, void* d_ws, size_t ws_size,
                              hipStream_t stream) {
  (void)in_sizes; (void)n_in; (void)out_size; (void)ws_size;
  const float* x      = (const float*)d_in[0];
  // d_in[1] = prefix_causal_mask (exactly tril -> implemented analytically)
  const float* W_attn = (const float*)d_in[2];
  const float* b_attn = (const float*)d_in[3];
  const float* W_proj = (const float*)d_in[4];
  const float* b_proj = (const float*)d_in[5];
  float* out = (float*)d_out;

  char* ws = (char*)d_ws;
  size_t off = 0;
  bf16_t* xb   = (bf16_t*)(ws + off); off += (size_t)Mrows * Cq * 2;        // 8 MiB
  bf16_t* wat  = (bf16_t*)(ws + off); off += (size_t)N1 * Cq * 2;           // 6 MiB
  bf16_t* wpt  = (bf16_t*)(ws + off); off += (size_t)Cq * Cq * 2;           // 2 MiB
  bf16_t* qws  = (bf16_t*)(ws + off); off += (size_t)Bq * Hq * Tq * Dq * 2; // 8 MiB
  bf16_t* kws  = (bf16_t*)(ws + off); off += (size_t)Bq * Hq * Tq * Dq * 2; // 8 MiB
  bf16_t* vtws = (bf16_t*)(ws + off); off += (size_t)Bq * Hq * Dq * Tq * 2; // 8 MiB
  bf16_t* yws  = (bf16_t*)(ws + off); off += (size_t)Mrows * Cq * 2;        // 8 MiB

  const int nx = Mrows * Cq;                      // 4,194,304
  csa_cvt_bf16<<<(nx + 255) / 256, 256, 0, stream>>>(x, xb, nx);
  csa_transpose_w<<<(N1 * Cq + 255) / 256, 256, 0, stream>>>(W_attn, wat, Cq, N1);
  csa_transpose_w<<<(Cq * Cq + 255) / 256, 256, 0, stream>>>(W_proj, wpt, Cq, Cq);

  // GEMM1: 32 m-blocks x 48 n-blocks, block = 128x64
  csa_gemm_qkv<<<32 * 48, 256, 0, stream>>>(xb, wat, b_attn, qws, kws, vtws);

  // B*H*(T/16) = 4096 wave-tiles, 8 per block
  csa_attention<<<4096 / 8, 256, 0, stream>>>(qws, kws, vtws, yws);

  // GEMM2: 32 m-blocks x 16 n-blocks
  csa_gemm_proj<<<32 * 16, 256, 0, stream>>>(yws, wpt, b_proj, out);
}
